// GraphWaveNet_42382737277243
// MI455X (gfx1250) — matrix operand order repaired and checked by reading, not compiled, not run on hardware
//
#include <hip/hip_runtime.h>
#include <hip/hip_bf16.h>
#include <math.h>

// ---- model dims ----
#define B_    64
#define N_    512
#define T0    13
#define RCc   32
#define DCc   32
#define SCc   256
#define ECc   512
#define OUTC  12
#define LYRS  8
#define GCIN  224   // (2*3+1)*DC

typedef __attribute__((ext_vector_type(16))) __bf16 v16bf;
typedef __attribute__((ext_vector_type(8)))  float  v8f;

static __device__ __forceinline__ unsigned short f2bf(float f) {
    union { float f; unsigned u; } x; x.f = f;
    unsigned u = x.u;
    unsigned r = u + 0x7FFFu + ((u >> 16) & 1u);   // round-to-nearest-even
    return (unsigned short)(r >> 16);
}

// ---------------- adp = softmax(relu(nv1 @ nv2), axis=1) ----------------
__global__ void k_adp(const float* __restrict__ nv1, const float* __restrict__ nv2,
                      float* __restrict__ adp) {
    int row = blockIdx.x;          // 512 rows
    int j   = threadIdx.x;         // 512 cols
    float acc = 0.f;
    #pragma unroll
    for (int k = 0; k < 10; ++k) acc += nv1[row * 10 + k] * nv2[k * 512 + j];
    acc = fmaxf(acc, 0.f);
    __shared__ float sm[512];
    sm[j] = acc; __syncthreads();
    for (int s = 256; s > 0; s >>= 1) { if (j < s) sm[j] = fmaxf(sm[j], sm[j + s]); __syncthreads(); }
    float m = sm[0]; __syncthreads();
    float e = __expf(acc - m);
    sm[j] = e; __syncthreads();
    for (int s = 256; s > 0; s >>= 1) { if (j < s) sm[j] += sm[j + s]; __syncthreads(); }
    adp[row * 512 + j] = e / sm[0];
}

// ---------------- pack support [512,512] f32 -> WMMA B-fragment bf16 ----------------
// B fragment (32x16 bf16, K x N): lane L: col = L%16, kbase = (L<16)?0:8, e<8 -> K = kbase+e; e>=8 -> K = 16+kbase+(e-8)
__global__ void k_pack_B(const float* __restrict__ A, unsigned short* __restrict__ dst) {
    int p = blockIdx.x * blockDim.x + threadIdx.x;
    if (p >= 512 * 512) return;
    int e    = p & 15;
    int lane = (p >> 4) & 31;
    int tile = p >> 9;
    const int KT = 512 / 32;
    int nTile = tile / KT, kTile = tile % KT;
    int k = kTile * 32 + ((lane & 16) ? 8 : 0) + ((e < 8) ? e : (16 + (e - 8)));
    int n = nTile * 16 + (lane & 15);
    dst[p] = f2bf(A[k * 512 + n]);
}

// ---------------- pack X [B,32,512,Tl] f32 -> WMMA A-fragment bf16, M=B*32*Tl, K=512 ----
// A fragment (16x32 bf16): lane L: row = L%16, kbase=(L<16)?0:8;
//   e<8 -> K = kbase+e ; e>=8 -> K = 16+kbase+(e-8)
__global__ void k_pack_A(const float* __restrict__ X, unsigned short* __restrict__ dst, int Tl) {
    int M = B_ * DCc * Tl;
    int total = M * 512;
    int p = blockIdx.x * blockDim.x + threadIdx.x;
    if (p >= total) return;
    int e    = p & 15;
    int lane = (p >> 4) & 31;
    int tile = p >> 9;
    const int KT = 16;
    int mTile = tile / KT, kTile = tile % KT;
    int m = mTile * 16 + (lane & 15);
    int kbase = (lane & 16) ? 8 : 0;
    int k = kTile * 32 + ((e < 8) ? (kbase + e) : (16 + kbase + (e - 8)));
    int b = m / (DCc * Tl);
    int c = (m / Tl) % DCc;
    int t = m % Tl;
    dst[p] = f2bf(X[(((long)b * DCc + c) * N_ + k) * Tl + t]);
}

// ---------------- bf16 WMMA GEMM: C[M,512] = A[M,512] * B[512,512], K=512 -------------
// 4 waves / block; wave w computes rows of mTile = blockIdx.x*4+w over a 64-wide
// column panel (4 n-tiles) whose B fragments are staged once in LDS (64 KB).
#define NTB 4   // n-tiles per block
__global__ void __launch_bounds__(128) k_wmma_gemm(const unsigned short* __restrict__ pA,
                                                   const unsigned short* __restrict__ pB,
                                                   float* __restrict__ out, int Tl) {
    const int KT = 16;                        // 512 / 32
    int wave   = threadIdx.x >> 5;
    int lane   = threadIdx.x & 31;
    int mTile  = blockIdx.x * 4 + wave;
    int nTile0 = blockIdx.y * NTB;

    __shared__ unsigned short sB[NTB * KT * 512];   // 64 KB B panel
    {
        const uint4* src = (const uint4*)(pB + (size_t)nTile0 * KT * 512);
        uint4*       dst = (uint4*)sB;
        const int total = NTB * KT * 512 / 8;       // uint4 = 8 bf16
        for (int i = threadIdx.x; i < total; i += 128) dst[i] = src[i];
    }
    __syncthreads();

    const v16bf* Ap = (const v16bf*)(pA + (size_t)mTile * KT * 512);
    const v16bf* Bs = (const v16bf*)sB;
    v8f acc[NTB] = {};
    for (int kt = 0; kt < KT; ++kt) {
        v16bf a = Ap[kt * 32 + lane];
        if (kt + 1 < KT) __builtin_prefetch((const void*)(Ap + (kt + 1) * 32 + lane), 0, 0);
        #pragma unroll
        for (int j = 0; j < NTB; ++j) {
            v16bf b = Bs[(j * KT + kt) * 32 + lane];
            acc[j] = __builtin_amdgcn_wmma_f32_16x16x32_bf16(false, a, false, b,
                                                             (short)0, acc[j], false, false);
        }
    }

    // C layout: VGPR v: row = mTile*16 + v + (lane<16?0:8), col = nTile*16 + lane%16
    int rbase = mTile * 16 + ((lane & 16) ? 8 : 0);
    #pragma unroll
    for (int j = 0; j < NTB; ++j) {
        int n = (nTile0 + j) * 16 + (lane & 15);
        #pragma unroll
        for (int v = 0; v < 8; ++v) {
            int m = rbase + v;
            int b = m / (DCc * Tl);
            int c = (m / Tl) % DCc;
            int t = m % Tl;
            out[(((long)b * DCc + c) * N_ + n) * Tl + t] = acc[j][v];
        }
    }
}

// ---------------- start 1x1 conv: [B,2,N,T] -> [B,32,N,T] ----------------
__global__ void k_start(const float* __restrict__ in, const float* __restrict__ W,
                        const float* __restrict__ bb, float* __restrict__ X) {
    int idx = blockIdx.x * blockDim.x + threadIdx.x;
    int total = B_ * RCc * N_ * T0;
    if (idx >= total) return;
    int t = idx % T0; int r = idx / T0;
    int v = r % N_;  r /= N_;
    int o = r % RCc; int b = r / RCc;
    float x0 = in[(((long)b * 2 + 0) * N_ + v) * T0 + t];
    float x1 = in[(((long)b * 2 + 1) * N_ + v) * T0 + t];
    X[idx] = bb[o] + W[o * 2 + 0] * x0 + W[o * 2 + 1] * x1;
}

// ---------------- gated dilated conv: tanh(f)*sigmoid(g) ----------------
__global__ void k_gated(const float* __restrict__ X,
                        const float* __restrict__ fW, const float* __restrict__ fb,
                        const float* __restrict__ gW, const float* __restrict__ gb,
                        float* __restrict__ out, int Tin, int d) {
    int Tout = Tin - d;
    int idx = blockIdx.x * blockDim.x + threadIdx.x;
    int total = B_ * DCc * N_ * Tout;
    if (idx >= total) return;
    int t = idx % Tout; int r = idx / Tout;
    int v = r % N_;  r /= N_;
    int o = r % DCc; int b = r / DCc;
    float f = fb[o], g = gb[o];
    #pragma unroll 8
    for (int ci = 0; ci < RCc; ++ci) {
        long base = (((long)b * RCc + ci) * N_ + v) * Tin + t;
        float x0 = X[base];
        float x1 = X[base + d];
        f += fW[(o * RCc + ci) * 2 + 0] * x0 + fW[(o * RCc + ci) * 2 + 1] * x1;
        g += gW[(o * RCc + ci) * 2 + 0] * x0 + gW[(o * RCc + ci) * 2 + 1] * x1;
    }
    out[(((long)b * DCc + o) * N_ + v) * Tout + t] = tanhf(f) * (1.f / (1.f + __expf(-g)));
}

// ---------------- skip accumulate (only last time slice matters) ----------------
__global__ void k_skip(const float* __restrict__ xg, const float* __restrict__ sW,
                       const float* __restrict__ sb, float* __restrict__ skip, int Tout) {
    int idx = blockIdx.x * blockDim.x + threadIdx.x;
    int total = B_ * SCc * N_;
    if (idx >= total) return;
    int v = idx % N_; int r = idx / N_;
    int o = r % SCc;  int b = r / SCc;
    float s = sb[o];
    #pragma unroll 8
    for (int c = 0; c < DCc; ++c)
        s += sW[o * DCc + c] * xg[(((long)b * DCc + c) * N_ + v) * Tout + (Tout - 1)];
    skip[idx] += s;
}

// ---------------- init next-x: residual (last Tout) + gconv bias ----------------
__global__ void k_init_next(const float* __restrict__ resid, const float* __restrict__ gb,
                            float* __restrict__ out, int Tin, int Tout) {
    int idx = blockIdx.x * blockDim.x + threadIdx.x;
    int total = B_ * DCc * N_ * Tout;
    if (idx >= total) return;
    int t = idx % Tout; int r = idx / Tout;
    int v = r % N_;  r /= N_;
    int o = r % DCc; int b = r / DCc;
    out[idx] = gb[o] + resid[(((long)b * DCc + o) * N_ + v) * Tin + (t + Tin - Tout)];
}

// ---------------- 1x1-conv accumulate for a 32-channel slice of h ----------------
__global__ void k_accum_slice(const float* __restrict__ Y, const float* __restrict__ W,
                              int cofs, float* __restrict__ out, int Tl) {
    int idx = blockIdx.x * blockDim.x + threadIdx.x;
    int total = B_ * DCc * N_ * Tl;
    if (idx >= total) return;
    int t = idx % Tl; int r = idx / Tl;
    int v = r % N_;  r /= N_;
    int o = r % DCc; int b = r / DCc;
    float s = 0.f;
    #pragma unroll 8
    for (int ci = 0; ci < 32; ++ci)
        s += W[o * GCIN + cofs + ci] * Y[(((long)b * DCc + ci) * N_ + v) * Tl + t];
    out[idx] += s;
}

// ---------------- batchnorm stats + apply ----------------
__global__ void k_bn_stats(const float* __restrict__ X, float* __restrict__ stats, int Tl) {
    int c = blockIdx.x;                       // 32 channels
    int tid = threadIdx.x;                    // 256
    long cnt = (long)B_ * N_ * Tl;
    float s = 0.f, s2 = 0.f;
    for (long i = tid; i < cnt; i += blockDim.x) {
        int t = (int)(i % Tl); long r = i / Tl;
        int v = (int)(r % N_); int b = (int)(r / N_);
        float x = X[(((long)b * DCc + c) * N_ + v) * Tl + t];
        s += x; s2 += x * x;
    }
    __shared__ float ss[256], ss2[256];
    ss[tid] = s; ss2[tid] = s2; __syncthreads();
    for (int k = 128; k > 0; k >>= 1) {
        if (tid < k) { ss[tid] += ss[tid + k]; ss2[tid] += ss2[tid + k]; }
        __syncthreads();
    }
    if (tid == 0) {
        float m = ss[0] / (float)cnt;
        stats[c * 2 + 0] = m;
        stats[c * 2 + 1] = ss2[0] / (float)cnt - m * m;
    }
}

__global__ void k_bn_apply(float* __restrict__ X, const float* __restrict__ stats,
                           const float* __restrict__ gamma, const float* __restrict__ beta,
                           int Tl) {
    int idx = blockIdx.x * blockDim.x + threadIdx.x;
    int total = B_ * DCc * N_ * Tl;
    if (idx >= total) return;
    int c = (idx / (N_ * Tl)) % DCc;
    float m = stats[c * 2], vv = stats[c * 2 + 1];
    X[idx] = (X[idx] - m) * rsqrtf(vv + 1e-5f) * gamma[c] + beta[c];
}

// ---------------- end MLP: relu -> 512 -> relu -> 12, one block per (b,node) -------
__global__ void k_end(const float* __restrict__ skip,
                      const float* __restrict__ e1W, const float* __restrict__ e1b,
                      const float* __restrict__ e2W, const float* __restrict__ e2b,
                      float* __restrict__ out) {
    int bv = blockIdx.x;
    int b = bv / N_, v = bv % N_;
    int tid = threadIdx.x;                    // 256
    __shared__ float s0[SCc];
    __shared__ float h1[ECc];
    s0[tid] = fmaxf(skip[((long)b * SCc + tid) * N_ + v], 0.f);
    __syncthreads();
    for (int e = tid; e < ECc; e += 256) {
        float a = e1b[e];
        #pragma unroll 8
        for (int c = 0; c < SCc; ++c) a += e1W[e * SCc + c] * s0[c];
        h1[e] = fmaxf(a, 0.f);
    }
    __syncthreads();
    if (tid < OUTC) {
        float a = e2b[tid];
        for (int e = 0; e < ECc; ++e) a += e2W[tid * ECc + e] * h1[e];
        out[((long)b * OUTC + tid) * N_ + v] = a;
    }
}

__global__ void k_zero(float* __restrict__ p, int n) {
    int i = blockIdx.x * blockDim.x + threadIdx.x;
    if (i < n) p[i] = 0.f;
}

// =====================================================================
extern "C" void kernel_launch(void* const* d_in, const int* in_sizes, int n_in,
                              void* d_out, int out_size, void* d_ws, size_t ws_size,
                              hipStream_t stream) {
    const float* in_x   = (const float*)d_in[0];
    const float* sup_a  = (const float*)d_in[1];
    const float* sup_b  = (const float*)d_in[2];
    const float* nv1    = (const float*)d_in[3];
    const float* nv2    = (const float*)d_in[4];
    const float* startW = (const float*)d_in[5];
    const float* startB = (const float*)d_in[6];
    const float* filtW  = (const float*)d_in[7];
    const float* filtB  = (const float*)d_in[8];
    const float* gateW  = (const float*)d_in[9];
    const float* gateB  = (const float*)d_in[10];
    const float* skipW  = (const float*)d_in[11];
    const float* skipB  = (const float*)d_in[12];
    const float* gcW    = (const float*)d_in[13];
    const float* gcB    = (const float*)d_in[14];
    const float* bnG    = (const float*)d_in[15];
    const float* bnBt   = (const float*)d_in[16];
    const float* e1W    = (const float*)d_in[17];
    const float* e1B    = (const float*)d_in[18];
    const float* e2W    = (const float*)d_in[19];
    const float* e2B    = (const float*)d_in[20];

    // workspace carve-up (256B aligned)
    char* ws = (char*)d_ws;
    auto carve = [&](size_t bytes) { char* p = ws; ws += (bytes + 255) & ~(size_t)255; return p; };
    float*          adp   = (float*)carve((size_t)512 * 512 * 4);
    unsigned short* pB0   = (unsigned short*)carve((size_t)512 * 512 * 2);
    unsigned short* pB1   = (unsigned short*)carve((size_t)512 * 512 * 2);
    unsigned short* pB2   = (unsigned short*)carve((size_t)512 * 512 * 2);
    float*          xA    = (float*)carve((size_t)B_ * DCc * N_ * T0 * 4);
    float*          xB    = (float*)carve((size_t)B_ * DCc * N_ * T0 * 4);
    float*          xg    = (float*)carve((size_t)B_ * DCc * N_ * 12 * 4);
    float*          ybuf  = (float*)carve((size_t)B_ * DCc * N_ * 12 * 4);
    unsigned short* pAx   = (unsigned short*)carve((size_t)B_ * DCc * 12 * 512 * 2);
    unsigned short* pAy   = (unsigned short*)carve((size_t)B_ * DCc * 12 * 512 * 2);
    float*          skip  = (float*)carve((size_t)B_ * SCc * N_ * 4);
    float*          stats = (float*)carve((size_t)DCc * 2 * 4);

    const int TPB = 256;
    auto blk = [](int n, int t) { return (n + t - 1) / t; };

    // adaptive adjacency + support packing (bf16 B-fragments)
    k_adp<<<512, 512, 0, stream>>>(nv1, nv2, adp);
    k_pack_B<<<blk(512 * 512, TPB), TPB, 0, stream>>>(sup_a, pB0);
    k_pack_B<<<blk(512 * 512, TPB), TPB, 0, stream>>>(sup_b, pB1);
    k_pack_B<<<blk(512 * 512, TPB), TPB, 0, stream>>>(adp,   pB2);
    unsigned short* pBs[3] = { pB0, pB1, pB2 };

    // skip accumulator + start conv
    k_zero<<<blk(B_ * SCc * N_, TPB), TPB, 0, stream>>>(skip, B_ * SCc * N_);
    k_start<<<blk(B_ * RCc * N_ * T0, TPB), TPB, 0, stream>>>(in_x, startW, startB, xA);

    static const int DIL[LYRS] = { 1, 2, 1, 2, 1, 2, 1, 2 };
    float* xcur = xA;
    float* xnxt = xB;
    int Tin = T0;

    for (int i = 0; i < LYRS; ++i) {
        int d = DIL[i], Tout = Tin - d;
        int nel = B_ * DCc * N_ * Tout;
        const float* fW = filtW + (size_t)i * DCc * RCc * 2;
        const float* fb = filtB + (size_t)i * DCc;
        const float* gW = gateW + (size_t)i * DCc * RCc * 2;
        const float* gb = gateB + (size_t)i * DCc;
        const float* sW = skipW + (size_t)i * SCc * DCc;
        const float* sb = skipB + (size_t)i * SCc;
        const float* cW = gcW   + (size_t)i * RCc * GCIN;
        const float* cB = gcB   + (size_t)i * RCc;

        k_gated<<<blk(nel, TPB), TPB, 0, stream>>>(xcur, fW, fb, gW, gb, xg, Tin, d);
        k_skip<<<blk(B_ * SCc * N_, TPB), TPB, 0, stream>>>(xg, sW, sb, skip, Tout);

        // xnxt = residual(last Tout) + gconv bias, then accumulate 7 conv1x1 slices
        k_init_next<<<blk(nel, TPB), TPB, 0, stream>>>(xcur, cB, xnxt, Tin, Tout);
        k_accum_slice<<<blk(nel, TPB), TPB, 0, stream>>>(xg, cW, 0, xnxt, Tout);

        // graph diffusion (WMMA bf16 GEMMs, K=512)
        int M  = B_ * DCc * Tout;
        int MT = M / 16;
        dim3 gemmGrid(MT / 4, 32 / NTB);
        k_pack_A<<<blk(M * 512, TPB), TPB, 0, stream>>>(xg, pAx, Tout);
        for (int s = 0; s < 3; ++s) {
            k_wmma_gemm<<<gemmGrid, 128, 0, stream>>>(pAx, pBs[s], ybuf, Tout);
            k_accum_slice<<<blk(nel, TPB), TPB, 0, stream>>>(ybuf, cW, 32 + s * 64, xnxt, Tout);
            k_pack_A<<<blk(M * 512, TPB), TPB, 0, stream>>>(ybuf, pAy, Tout);
            k_wmma_gemm<<<gemmGrid, 128, 0, stream>>>(pAy, pBs[s], ybuf, Tout);
            k_accum_slice<<<blk(nel, TPB), TPB, 0, stream>>>(ybuf, cW, 64 + s * 64, xnxt, Tout);
        }

        // batchnorm in place
        k_bn_stats<<<32, 256, 0, stream>>>(xnxt, stats, Tout);
        k_bn_apply<<<blk(nel, TPB), TPB, 0, stream>>>(xnxt, stats,
                                                      bnG + (size_t)i * RCc,
                                                      bnBt + (size_t)i * RCc, Tout);
        float* tmp = xcur; xcur = xnxt; xnxt = tmp;
        Tin = Tout;
    }

    // end MLP -> d_out [B, 12, N, 1]
    k_end<<<B_ * N_, 256, 0, stream>>>(skip, e1W, e1B, e2W, e2B, (float*)d_out);
    (void)in_sizes; (void)n_in; (void)out_size; (void)ws_size;
}
